// Encoder_63333587747633
// MI455X (gfx1250) — compile-verified, matrix operand order checked by reading, not run on hardware
//
#include <hip/hip_runtime.h>

// ---------------------------------------------------------------------------
// 2-layer GRU (Keras reset_after, relu candidate), B=64, T=2048, F=15, U=256.
// Only final states are needed -> persistent kernel, both layers pipelined,
// grid barrier per timestep. Recurrent GEMMs use v_wmma_f32_16x16x32_bf16
// with fp32 accumulation; the fp32 master state lives in registers per lane
// (lane ownership matches the WMMA 16x16 f32 C-layout). Recurrent weight
// slices are staged in LDS (ds_load_b128 feeds), W2^T streams from L2
// (384 KB -> permanently hot), h-state exchange (32 KB bf16) stays in L2.
// ---------------------------------------------------------------------------

typedef __bf16 bf16_t;
typedef __attribute__((ext_vector_type(16))) __bf16 bf16x16;
typedef __attribute__((ext_vector_type(8)))  __bf16 bf16x8;
typedef __attribute__((ext_vector_type(8)))  float  f32x8;

#define NWG    16
#define NTHR   256
#define BATCH  64
#define UNITS  256
#define SEQT   2048
#define FEAT   15
#define HBUF_ELEMS (BATCH*UNITS)   // 16384 elements per state buffer

__device__ __forceinline__ bf16x8 ld8(const bf16_t* p){ return *(const bf16x8*)p; }
__device__ __forceinline__ bf16x16 cat8(bf16x8 a, bf16x8 b){
  return __builtin_shufflevector(a, b, 0,1,2,3,4,5,6,7,8,9,10,11,12,13,14,15);
}
__device__ __forceinline__ f32x8 wmma_bf16(bf16x16 a, bf16x16 b, f32x8 c){
  // D = A(16x32 bf16) * B(32x16 bf16) + C(16x16 f32)
  return __builtin_amdgcn_wmma_f32_16x16x32_bf16(false, a, false, b, (short)0, c, false, false);
}
__device__ __forceinline__ float sigmoidf_(float x){ return 1.0f/(1.0f + __expf(-x)); }

// ---------------------------------------------------------------------------
// Prep: transpose U1/W2/U2 to [768][256] bf16 (so every WMMA B-operand lane
// reads two contiguous 16B chunks), zero the state double-buffers + barrier.
// Runs every launch -> deterministic (workspace is poisoned by the harness).
// ---------------------------------------------------------------------------
__global__ void prep_kernel(const float* __restrict__ U1, const float* __restrict__ W2,
                            const float* __restrict__ U2,
                            bf16_t* __restrict__ U1T, bf16_t* __restrict__ W2T,
                            bf16_t* __restrict__ U2T,
                            bf16_t* __restrict__ s1buf, bf16_t* __restrict__ h2buf,
                            unsigned* __restrict__ bar)
{
  const int gid = blockIdx.x*blockDim.x + threadIdx.x;
  const int nth = gridDim.x*blockDim.x;
  const int NE  = 768*256;
  for (int i = gid; i < NE; i += nth){
    int n = i >> 8, k = i & 255;            // T[n][k] = M[k][n]
    U1T[i] = (bf16_t)U1[k*768 + n];
    W2T[i] = (bf16_t)W2[k*768 + n];
    U2T[i] = (bf16_t)U2[k*768 + n];
  }
  for (int i = gid; i < 2*HBUF_ELEMS; i += nth){
    s1buf[i] = (bf16_t)0.0f;
    h2buf[i] = (bf16_t)0.0f;
  }
  if (gid == 0) *bar = 0u;
}

// ---------------------------------------------------------------------------
// Persistent kernel: 16 WGs x 256 threads (8 wave32 each).
//   WG 0..7  -> layer 1, units [32*wg, 32*wg+32)
//   WG 8..15 -> layer 2, units [32*(wg-8), ...)
// Wave w: M-tile = w&3 (16 batch rows), unit-tile = w>>2 (16 units).
// Iteration i: layer1 computes h1[i] (i<T); layer2 computes h2[i-1] (i>=1).
// State buffers are double-buffered bf16 in workspace (L2-resident).
// ---------------------------------------------------------------------------
__launch_bounds__(NTHR, 1)
__global__ void gru_persistent(const float* __restrict__ x,
                               const float* __restrict__ W1,
                               const float* __restrict__ b1,
                               const float* __restrict__ b2,
                               const bf16_t* __restrict__ U1T,
                               const bf16_t* __restrict__ W2T,
                               const bf16_t* __restrict__ U2T,
                               bf16_t* __restrict__ s1buf,
                               bf16_t* __restrict__ h2buf,
                               unsigned* __restrict__ bar,
                               float* __restrict__ out)
{
  __shared__ __align__(16) bf16_t sU[3*32*256];   // 48 KB recurrent-weight slice
  __shared__ __align__(16) float  sX[BATCH*16];   // 4 KB x_t staging (layer 1)

  const int tid   = threadIdx.x;
  const int wg    = blockIdx.x;
  const bool isL2 = (wg >= 8);
  const int u0    = (wg & 7)*32;        // first unit owned by this WG

  const int wv   = tid >> 5;            // wave id 0..7
  const int lane = tid & 31;
  const int m0   = (wv & 3)*16;         // batch-row tile base
  const int ut16 = (wv >> 2)*16;        // unit-tile base within WG slice
  const int l16  = lane & 15;
  const int h8   = (lane >> 4)*8;       // K half-offset (A/B) == M half-offset (C)
  const int u    = u0 + ut16 + l16;     // this lane's unit column

  // ---- stage recurrent weight slice [gate][32 units][256 k] into LDS ----
  {
    const bf16_t* UT = isL2 ? U2T : U1T;
    for (int idx = tid; idx < (3*32*256)/8; idx += NTHR){
      int e = idx*8; int k = e & 255; int row = e >> 8;   // row 0..95
      int g = row >> 5, ul = row & 31;
      *(bf16x8*)(sU + e) = ld8(UT + (g*256 + u0 + ul)*256 + k);
    }
  }

  // ---- per-lane constants hoisted out of the time loop ----
  float binv[3], brecv[3];
  float wc[3][FEAT];                    // W1 columns (layer 1 only)
  {
    const float* bb = isL2 ? b2 : b1;
    #pragma unroll
    for (int g = 0; g < 3; ++g){
      const int col = g*256 + u;
      binv[g]  = bb[col];
      brecv[g] = bb[768 + col];
    }
    if (!isL2){
      #pragma unroll
      for (int g = 0; g < 3; ++g){
        const int col = g*256 + u;
        for (int f = 0; f < FEAT; ++f) wc[g][f] = W1[f*768 + col];
      }
    }
  }

  float hreg[8];                        // fp32 master state, lane-owned (C layout)
  #pragma unroll
  for (int v = 0; v < 8; ++v) hreg[v] = 0.0f;

  __syncthreads();

  for (int i = 0; i <= SEQT; ++i){
    if (!isL2 && i < SEQT){
      // ================= layer 1, timestep t = i =================
      const int t = i;
      f32x8 az = {}, ar = {}, ah = {};
      const bf16_t* Arow = s1buf + ((t+1)&1)*HBUF_ELEMS + (m0 + l16)*UNITS + h8;
      const bf16_t* Bb   = sU + (ut16 + l16)*256 + h8;
      #pragma unroll
      for (int kt = 0; kt < 8; ++kt){
        const int k0 = kt*32;
        bf16x16 a  = cat8(ld8(Arow + k0),            ld8(Arow + k0 + 16));
        bf16x16 bz = cat8(ld8(Bb + 0*8192 + k0),     ld8(Bb + 0*8192 + k0 + 16));
        bf16x16 br = cat8(ld8(Bb + 1*8192 + k0),     ld8(Bb + 1*8192 + k0 + 16));
        bf16x16 bh = cat8(ld8(Bb + 2*8192 + k0),     ld8(Bb + 2*8192 + k0 + 16));
        az = wmma_bf16(a, bz, az);
        ar = wmma_bf16(a, br, ar);
        ah = wmma_bf16(a, bh, ah);
      }
      // x_t -> LDS, then xp = x_t @ W1 + b_in on VALU (K = 15)
      for (int idx = tid; idx < BATCH*FEAT; idx += NTHR){
        int b = idx/FEAT, f = idx - b*FEAT;
        sX[b*16 + f] = x[(b*SEQT + t)*FEAT + f];
      }
      __syncthreads();
      #pragma unroll
      for (int v = 0; v < 8; ++v){
        const int m = m0 + h8 + v;      // C layout: VGPR v -> M = v (+8 for hi lanes)
        float xr[FEAT];
        #pragma unroll
        for (int f = 0; f < FEAT; ++f) xr[f] = sX[m*16 + f];
        float xpz = binv[0], xpr = binv[1], xph = binv[2];
        #pragma unroll
        for (int f = 0; f < FEAT; ++f){
          xpz = fmaf(xr[f], wc[0][f], xpz);
          xpr = fmaf(xr[f], wc[1][f], xpr);
          xph = fmaf(xr[f], wc[2][f], xph);
        }
        const float z  = sigmoidf_(xpz + az[v] + brecv[0]);
        const float r  = sigmoidf_(xpr + ar[v] + brecv[1]);
        const float hh = fmaxf(0.0f, xph + r*(ah[v] + brecv[2]));
        const float hn = z*hreg[v] + (1.0f - z)*hh;
        hreg[v] = hn;
        s1buf[(t&1)*HBUF_ELEMS + m*UNITS + u] = (bf16_t)hn;
      }
    } else if (isL2 && i >= 1){
      // ================= layer 2, timestep t = i-1 =================
      const int t = i - 1;
      f32x8 az = {}, ar = {}, arh = {}, axh = {};
      const bf16_t* A2 = h2buf + ((t+1)&1)*HBUF_ELEMS + (m0 + l16)*UNITS + h8;
      const bf16_t* A1 = s1buf + (t&1)*HBUF_ELEMS     + (m0 + l16)*UNITS + h8;
      const bf16_t* Bu = sU  + (ut16 + l16)*256 + h8;   // U2^T slice in LDS
      const bf16_t* Bw = W2T + u*256 + h8;              // W2^T from L2
      #pragma unroll
      for (int kt = 0; kt < 8; ++kt){
        const int k0 = kt*32;
        bf16x16 a2 = cat8(ld8(A2 + k0),              ld8(A2 + k0 + 16));
        bf16x16 uz = cat8(ld8(Bu + 0*8192  + k0),    ld8(Bu + 0*8192  + k0 + 16));
        bf16x16 ur = cat8(ld8(Bu + 1*8192  + k0),    ld8(Bu + 1*8192  + k0 + 16));
        bf16x16 uh = cat8(ld8(Bu + 2*8192  + k0),    ld8(Bu + 2*8192  + k0 + 16));
        az  = wmma_bf16(a2, uz, az);
        ar  = wmma_bf16(a2, ur, ar);
        arh = wmma_bf16(a2, uh, arh);
        bf16x16 a1 = cat8(ld8(A1 + k0),              ld8(A1 + k0 + 16));
        bf16x16 wz = cat8(ld8(Bw + 0*65536 + k0),    ld8(Bw + 0*65536 + k0 + 16));
        bf16x16 wr = cat8(ld8(Bw + 1*65536 + k0),    ld8(Bw + 1*65536 + k0 + 16));
        bf16x16 wh = cat8(ld8(Bw + 2*65536 + k0),    ld8(Bw + 2*65536 + k0 + 16));
        az  = wmma_bf16(a1, wz, az);    // z: x-proj + rec fused in one acc
        ar  = wmma_bf16(a1, wr, ar);    // r: fused
        axh = wmma_bf16(a1, wh, axh);   // h: x-proj kept separate (r scales rec only)
      }
      #pragma unroll
      for (int v = 0; v < 8; ++v){
        const int m = m0 + h8 + v;
        const float z  = sigmoidf_(az[v] + binv[0] + brecv[0]);
        const float r  = sigmoidf_(ar[v] + binv[1] + brecv[1]);
        const float hh = fmaxf(0.0f, axh[v] + binv[2] + r*(arh[v] + brecv[2]));
        const float hn = z*hreg[v] + (1.0f - z)*hh;
        hreg[v] = hn;
        h2buf[(t&1)*HBUF_ELEMS + m*UNITS + u] = (bf16_t)hn;
      }
    }

    // -------- grid barrier (monotonic counter; 16 WGs are co-resident) --------
    __threadfence();
    __syncthreads();
    if (tid == 0){
      __hip_atomic_fetch_add(bar, 1u, __ATOMIC_ACQ_REL, __HIP_MEMORY_SCOPE_AGENT);
      const unsigned tgt = (unsigned)((i + 1) * NWG);
      while (__hip_atomic_load(bar, __ATOMIC_ACQUIRE, __HIP_MEMORY_SCOPE_AGENT) < tgt)
        __builtin_amdgcn_s_sleep(8);
    }
    __syncthreads();
  }

  // -------- final outputs: (x = h2_final, state1 = h1_final, state2 = h2_final)
  #pragma unroll
  for (int v = 0; v < 8; ++v){
    const int m = m0 + h8 + v;
    if (!isL2){
      out[HBUF_ELEMS + m*UNITS + u] = hreg[v];          // state1
    } else {
      out[m*UNITS + u]                = hreg[v];        // x (= seq2[:,-1,:])
      out[2*HBUF_ELEMS + m*UNITS + u] = hreg[v];        // state2
    }
  }
}

// ---------------------------------------------------------------------------
extern "C" void kernel_launch(void* const* d_in, const int* in_sizes, int n_in,
                              void* d_out, int out_size, void* d_ws, size_t ws_size,
                              hipStream_t stream)
{
  (void)in_sizes; (void)n_in; (void)out_size;
  const float* x  = (const float*)d_in[0];   // (64, 2048, 15)
  const float* W1 = (const float*)d_in[1];   // (15, 768)
  const float* U1 = (const float*)d_in[2];   // (256, 768)
  const float* b1 = (const float*)d_in[3];   // (2, 768)
  const float* W2 = (const float*)d_in[4];   // (256, 768)
  const float* U2 = (const float*)d_in[5];   // (256, 768)
  const float* b2 = (const float*)d_in[6];   // (2, 768)
  float* out = (float*)d_out;                // 3 * 64*256 fp32

  // Workspace layout (~1.28 MB total, assumed <= ws_size):
  char* ws = (char*)d_ws;
  unsigned* bar  = (unsigned*)(ws);                                  // 256 B
  bf16_t*  s1buf = (bf16_t*)(ws + 256);                              // 64 KB (2x h1 bf16)
  bf16_t*  h2buf = (bf16_t*)(ws + 256 + 65536);                      // 64 KB (2x h2 bf16)
  bf16_t*  U1T   = (bf16_t*)(ws + 256 + 2*65536);                    // 384 KB
  bf16_t*  W2T   = (bf16_t*)(ws + 256 + 2*65536 + 393216);           // 384 KB
  bf16_t*  U2T   = (bf16_t*)(ws + 256 + 2*65536 + 2*393216);         // 384 KB
  (void)ws_size;

  prep_kernel<<<256, NTHR, 0, stream>>>(U1, W2, U2, U1T, W2T, U2T, s1buf, h2buf, bar);
  gru_persistent<<<NWG, NTHR, 0, stream>>>(x, W1, b1, b2, U1T, W2T, U2T,
                                           s1buf, h2buf, bar, out);
}
